// AttentionLayer_87531433492672
// MI455X (gfx1250) — matrix operand order repaired and checked by reading, not compiled, not run on hardware
//
#include <hip/hip_runtime.h>
#include <hip/hip_bf16.h>

// ---------------------------------------------------------------------------
// Additive attention:  out = softmax_S( tanh(entities@W1 + contexts@W2) @ v ) * contexts
// B=32, S=2048, D=1024, H=1024 (derived from in_sizes at launch).
// Dominant cost: contexts@W2 (137 GFLOP) -> v_wmma_f32_16x16x32_bf16.
//   * w2 pre-packed to bf16 in per-lane fragment order (2MB, L2-resident)
//   * block tile 64(M) x 1024(N = full H): contexts streamed from HBM ONCE;
//     B L2-traffic 2MB*(S*B/64) = 2GB
//   * per wave: 4 m-tiles x 8 n-tiles = 32 independent WMMA accumulators
// ---------------------------------------------------------------------------

typedef __bf16 v16bf __attribute__((ext_vector_type(16)));
typedef float  v8f   __attribute__((ext_vector_type(8)));
typedef float  v4f   __attribute__((ext_vector_type(4)));

#define RE_BLOCK   256
#define SM_BLOCK   256
#define SC_WAVES   8                         // waves per workgroup
#define SC_MT      4                         // 16-row m-tiles per block (all waves)
#define SC_NT      8                         // 16-col n-tiles per wave
#define SC_MROWS   (SC_MT * 16)              // 64 S-rows per block
#define SC_NCOLS   (SC_WAVES * SC_NT * 16)   // 1024 H-cols per block (= H)

// ---------------------------------------------------------------------------
// Kernel 0: pack w2 [D,H] f32 row-major -> bf16 in WMMA B-fragment order.
// Fragment layout (16-bit B 32x16, wave32):
//   lane<16 holds K = kblk*32 + 0..7,16..23, lane>=16 holds K = kblk*32 + 8..15,24..31,
//   column n = ntile*16 + (lane&15).
// Packed index: (((nt * D/32) + kblk) * 32 + lane) * 16 + i
// => each lane's 16 values are 32 contiguous bytes; a fragment is 1KB coalesced.
// ---------------------------------------------------------------------------
__global__ __launch_bounds__(256)
void pack_w2_kernel(const float* __restrict__ w2,
                    __bf16* __restrict__ w2p, int D, int H)
{
    const int idx = blockIdx.x * 256 + threadIdx.x;     // over D*H
    if (idx >= D * H) return;
    const int i    = idx & 15;
    const int lane = (idx >> 4) & 31;
    const int rest = idx >> 9;
    const int kblks = D >> 5;
    const int kblk = rest % kblks;
    const int nt   = rest / kblks;
    const int k = kblk * 32 + ((lane < 16) ? 0 : 8) + (i & 7) + ((i & 8) << 1);
    const int n = nt * 16 + (lane & 15);
    w2p[idx] = (__bf16)w2[(size_t)k * H + n];
}

// ---------------------------------------------------------------------------
// Kernel 1: re[b,h] = entities[b,:] . w1_w[:,h] + w1_b[h]
// ---------------------------------------------------------------------------
__global__ __launch_bounds__(RE_BLOCK)
void re_kernel(const float* __restrict__ entities,
               const float* __restrict__ w1_w,
               const float* __restrict__ w1_b,
               float* __restrict__ re,
               int D, int H)
{
    __shared__ float e[2048];            // D <= 2048
    const int b = blockIdx.y;
    const int h = blockIdx.x * RE_BLOCK + threadIdx.x;

    for (int d = threadIdx.x; d < D; d += RE_BLOCK)
        e[d] = entities[(size_t)b * D + d];
    __syncthreads();

    float acc = w1_b[h];
    #pragma unroll 4
    for (int d = 0; d < D; ++d)
        acc = fmaf(e[d], w1_w[(size_t)d * H + h], acc);   // coalesced over h
    re[(size_t)b * H + h] = acc;
}

// ---------------------------------------------------------------------------
// Kernel 2 helper: zero the score accumulators (atomics accumulate into them)
// ---------------------------------------------------------------------------
__global__ void zero_kernel(float* __restrict__ p, int n)
{
    int i = blockIdx.x * blockDim.x + threadIdx.x;
    if (i < n) p[i] = 0.0f;
}

// ---------------------------------------------------------------------------
// Kernel 3: scores[b,s] += sum_h tanh(re[b,h] + rc[b,s,h] + b2[h]) * v_w[h]
// Block: 64 S-rows x full H (1024 cols), 8 waves. Each wave: 4 m-tiles
// (shared rows) x 8 n-tiles (own columns) = 32 WMMA accumulators.
// Per k-block: 16 A-b128 loads (f32->bf16) + 16 B-b128 loads (packed bf16)
// feed 32 v_wmma_f32_16x16x32_bf16 (1.0 b128 load per WMMA).
// ---------------------------------------------------------------------------
__global__ __launch_bounds__(SC_WAVES * 32)
void scores_kernel(const float* __restrict__ contexts,   // [B,S,D]
                   const __bf16* __restrict__ w2p,       // packed [H/16][D/32][32][16]
                   const float* __restrict__ w2_b,       // [H]
                   const float* __restrict__ re,         // [B,H]
                   const float* __restrict__ v_w,        // [H]
                   float* __restrict__ scores,           // [B,S]
                   int S, int D, int H)
{
    const int b    = blockIdx.z;
    const int s0   = blockIdx.x * SC_MROWS;
    const int wave = threadIdx.x >> 5;
    const int lane = threadIdx.x & 31;
    const int nw0  = blockIdx.y * SC_NCOLS + wave * (SC_NT * 16); // first H col
    const int kblks = D >> 5;

    const int  mrow = lane & 15;          // A: row index; C: column index
    const bool hi   = lane >= 16;
    const int  ka   = hi ? 8 : 0;         // A fragment K offset (ISA layout)

    const float* __restrict__ arow[SC_MT];
    #pragma unroll
    for (int mt = 0; mt < SC_MT; ++mt)
        arow[mt] = contexts + ((size_t)b * S + (s0 + mt * 16 + mrow)) * D;

    // Per-lane base into packed B: tile stride = kblks*512 elems, kblk stride = 512.
    const __bf16* __restrict__ bbase =
        w2p + (((size_t)(nw0 >> 4) * kblks) * 32 + lane) * 16;

    v8f acc[SC_MT][SC_NT];
    #pragma unroll
    for (int mt = 0; mt < SC_MT; ++mt)
        #pragma unroll
        for (int t = 0; t < SC_NT; ++t)
            acc[mt][t] = (v8f){0.f, 0.f, 0.f, 0.f, 0.f, 0.f, 0.f, 0.f};

    for (int kblk = 0; kblk < kblks; ++kblk) {
        // ---- A fragments (one per m-tile), 16-bit A 16x32 layout:
        //   lane<16:  K = k0+0..7   and k0+16..23
        //   lane>=16: K = k0+8..15  and k0+24..31
        const int k0 = kblk * 32;
        v16bf afrag[SC_MT];
        #pragma unroll
        for (int mt = 0; mt < SC_MT; ++mt) {
            const v4f* pa = (const v4f*)(arow[mt] + k0 + ka);
            const v4f* pb = (const v4f*)(arow[mt] + k0 + ka + 16);
            v4f a0 = pa[0], a1 = pa[1];
            v4f a2 = pb[0], a3 = pb[1];
            #pragma unroll
            for (int i = 0; i < 4; ++i) {
                afrag[mt][i]      = (__bf16)a0[i];
                afrag[mt][4 + i]  = (__bf16)a1[i];
                afrag[mt][8 + i]  = (__bf16)a2[i];
                afrag[mt][12 + i] = (__bf16)a3[i];
            }
        }
        if (k0 + 64 < D) __builtin_prefetch(arow[0] + k0 + 64, 0, 1);

        // ---- B fragments: 32B/lane vector loads from packed array; each
        // fragment feeds 4 WMMAs (one per m-tile).
        #pragma unroll
        for (int t = 0; t < SC_NT; ++t) {
            const v16bf bfrag =
                *(const v16bf*)(bbase + ((size_t)t * kblks + kblk) * 512);
            #pragma unroll
            for (int mt = 0; mt < SC_MT; ++mt)
                acc[mt][t] = __builtin_amdgcn_wmma_f32_16x16x32_bf16(
                    false, afrag[mt], false, bfrag, (short)0, acc[mt][t],
                    false, false);
        }
    }

    // ---- epilogue: tanh(re + rc + b2) * v_w, reduce over N (16-lane group)
    // C layout: VGPR r -> row M = r + (hi?8:0), column N = lane&15.
    const float* __restrict__ re_row = re + (size_t)b * H;

    #pragma unroll
    for (int mt = 0; mt < SC_MT; ++mt) {
        float part[8];
        #pragma unroll
        for (int r = 0; r < 8; ++r) part[r] = 0.f;

        #pragma unroll
        for (int t = 0; t < SC_NT; ++t) {
            const int   n    = nw0 + t * 16 + mrow;
            const float addn = w2_b[n] + re_row[n];
            const float vw   = v_w[n];
            #pragma unroll
            for (int r = 0; r < 8; ++r)
                part[r] += tanhf(acc[mt][t][r] + addn) * vw;
        }

        #pragma unroll
        for (int r = 0; r < 8; ++r) {
            float p = part[r];
            p += __shfl_xor(p, 1, 32);
            p += __shfl_xor(p, 2, 32);
            p += __shfl_xor(p, 4, 32);
            p += __shfl_xor(p, 8, 32);
            if (mrow == 0) {
                const int m = r + (hi ? 8 : 0);
                atomicAdd(&scores[(size_t)b * S + s0 + mt * 16 + m], p);
            }
        }
    }
}

// ---------------------------------------------------------------------------
// Kernel 4: softmax over S per batch (v_b dropped: softmax is shift-invariant)
// ---------------------------------------------------------------------------
__global__ __launch_bounds__(SM_BLOCK)
void softmax_kernel(const float* __restrict__ scores,
                    float* __restrict__ weights, int S)
{
    __shared__ float red[SM_BLOCK];
    const int b   = blockIdx.x;
    const int tid = threadIdx.x;
    const float* __restrict__ sc = scores + (size_t)b * S;

    float m = -INFINITY;
    for (int s = tid; s < S; s += SM_BLOCK) m = fmaxf(m, sc[s]);
    red[tid] = m;
    __syncthreads();
    for (int o = SM_BLOCK / 2; o > 0; o >>= 1) {
        if (tid < o) red[tid] = fmaxf(red[tid], red[tid + o]);
        __syncthreads();
    }
    m = red[0];
    __syncthreads();

    float sum = 0.f;
    for (int s = tid; s < S; s += SM_BLOCK) sum += __expf(sc[s] - m);
    red[tid] = sum;
    __syncthreads();
    for (int o = SM_BLOCK / 2; o > 0; o >>= 1) {
        if (tid < o) red[tid] += red[tid + o];
        __syncthreads();
    }
    const float inv = 1.0f / red[0];

    float* __restrict__ w = weights + (size_t)b * S;
    for (int s = tid; s < S; s += SM_BLOCK)
        w[s] = __expf(sc[s] - m) * inv;
}

// ---------------------------------------------------------------------------
// Kernel 5: out[b,s,:] = weights[b,s] * contexts[b,s,:]  (streaming, 128-bit)
// One block per (b,s) row. NT stores: output never re-read.
// ---------------------------------------------------------------------------
__global__ __launch_bounds__(256)
void scale_kernel(const float* __restrict__ contexts,
                  const float* __restrict__ weights,
                  float* __restrict__ out, int D)
{
    const size_t row = blockIdx.x;              // b*S + s
    const float  w   = weights[row];
    const v4f* __restrict__ src = (const v4f*)(contexts + row * D);
    v4f*       __restrict__ dst = (v4f*)(out + row * D);
    const int d4 = D >> 2;
    for (int j = threadIdx.x; j < d4; j += 256) {
        v4f c = src[j];
        v4f o = c * w;
        __builtin_nontemporal_store(o, &dst[j]);
    }
}

// ---------------------------------------------------------------------------
extern "C" void kernel_launch(void* const* d_in, const int* in_sizes, int n_in,
                              void* d_out, int out_size, void* d_ws, size_t ws_size,
                              hipStream_t stream)
{
    const float* entities = (const float*)d_in[0];   // [B,D]
    const float* contexts = (const float*)d_in[1];   // [B,S,D]
    const float* w1_w     = (const float*)d_in[2];   // [D,H]
    const float* w1_b     = (const float*)d_in[3];   // [H]
    const float* w2_w     = (const float*)d_in[4];   // [D,H]
    const float* w2_b     = (const float*)d_in[5];   // [H]
    const float* v_w      = (const float*)d_in[6];   // [H,1]
    // d_in[7] = v_b : constant shift, cancelled by softmax.

    const int H = in_sizes[3];
    const int D = in_sizes[2] / H;
    const int B = in_sizes[0] / D;
    const int S = in_sizes[1] / (B * D);

    // Workspace carve (d_ws is poisoned once; everything below is fully
    // rewritten every call, so no cross-call state).
    __bf16* w2p    = (__bf16*)d_ws;                          // [D*H] bf16 (2MB)
    float*  re     = (float*)(w2p + (size_t)D * H);          // [B,H]
    float*  scores = re + (size_t)B * H;                     // [B,S]
    float*  weights= scores + (size_t)B * S;                 // [B,S]

    // 0) repack w2 into WMMA fragment order (bf16)
    pack_w2_kernel<<<(D * H + 255) / 256, 256, 0, stream>>>(w2_w, w2p, D, H);

    // 1) re = entities @ w1 + b1
    re_kernel<<<dim3(H / RE_BLOCK, B), RE_BLOCK, 0, stream>>>(
        entities, w1_w, w1_b, re, D, H);

    // 2) scores = 0, then accumulate WMMA tanh-attention partial sums
    zero_kernel<<<(B * S + 255) / 256, 256, 0, stream>>>(scores, B * S);

    scores_kernel<<<dim3(S / SC_MROWS, H / SC_NCOLS, B), SC_WAVES * 32, 0, stream>>>(
        contexts, w2p, w2_b, re, v_w, scores, S, D, H);

    // 3) softmax over S
    softmax_kernel<<<B, SM_BLOCK, 0, stream>>>(scores, weights, S);

    // 4) out = weights * contexts
    scale_kernel<<<(size_t)B * S, 256, 0, stream>>>(contexts, weights, (float*)d_out, D);
}
